// CrossNetMix_33139967656516
// MI455X (gfx1250) — matrix operand order repaired and checked by reading, not compiled, not run on hardware
//
#include <hip/hip_runtime.h>
#include <hip/hip_bf16.h>
#include <math.h>

// ---------------------------------------------------------------------------
// CrossNetMix on gfx1250: fully fused 3-layer kernel, bf16 WMMA, fp32 accum.
//   stage1: T = x @ Vaug^T   (Vaug packed [272 x 1024] bf16, cols 256..259 = gating)
//   stage2: W = v @ Cblk^T   (block-diagonal C packed [256 x 256] bf16)
//   stage3: Y = wg @ Upk^T   (U packed [1024 x 256] bf16)
//   x <- x + x0 * (bias + Y)
// Each wave owns 16 rows through all layers; no inter-wave sync required.
// All GEMM chunk widths are compile-time constants -> branch-free K loops.
// ---------------------------------------------------------------------------

typedef __attribute__((ext_vector_type(16))) __bf16 bf16x16;
typedef __attribute__((ext_vector_type(8)))  __bf16 bf16x8;
typedef __attribute__((ext_vector_type(8)))  float  f32x8;
typedef __attribute__((ext_vector_type(4)))  float  f32x4;

#define Lc    3
#define Ec    4
#define Dd    1024
#define Rr    64
#define Bb    32768
#define NAUG  272      // E*R + 16 (4 gating cols + 12 zero pad) -> 17 N-tiles
#define NKR   256      // E*R
#define SV    264      // LDS row stride (bf16) for v/wg buffers (bank-spread pad)
#define WAVES 8
#define TM    16

static __device__ inline bf16x16 cat8(bf16x8 lo, bf16x8 hi) {
  return __builtin_shufflevector(lo, hi, 0,1,2,3,4,5,6,7,8,9,10,11,12,13,14,15);
}

// A-fragment (16x32 bf16) from row-major fp32 in global memory.
// lane: row = lane&15, khalf = (lane>>4)*8; elems 0..7 = K khalf..khalf+7,
// elems 8..15 = K khalf+16..khalf+23.  p = base + row*ldA + kbase + khalf.
static __device__ inline bf16x16 load_a_f32(const float* p) {
  f32x4 a0 = *(const f32x4*)(p + 0);
  f32x4 a1 = *(const f32x4*)(p + 4);
  f32x4 a2 = *(const f32x4*)(p + 16);
  f32x4 a3 = *(const f32x4*)(p + 20);
  bf16x16 r;
#pragma unroll
  for (int i = 0; i < 4; ++i) {
    r[i]      = (__bf16)a0[i];
    r[4 + i]  = (__bf16)a1[i];
    r[8 + i]  = (__bf16)a2[i];
    r[12 + i] = (__bf16)a3[i];
  }
  return r;
}

// A-fragment from row-major bf16 in LDS. p = buf + row*SV + kbase + khalf.
static __device__ inline bf16x16 load_a_lds(const __bf16* p) {
  bf16x8 lo = *(const bf16x8*)(p);
  bf16x8 hi = *(const bf16x8*)(p + 16);
  return cat8(lo, hi);
}

// B-fragment (32x16 bf16) from weight packed as [N x K] row-major bf16.
// lane: col = lane&15, khalf = (lane>>4)*8; elems 0..7 = K khalf..khalf+7,
// elems 8..15 = K khalf+16..khalf+23.  p = base + col*ldB + kbase + khalf.
static __device__ inline bf16x16 load_b(const __bf16* p) {
  bf16x8 lo = *(const bf16x8*)(p);
  bf16x8 hi = *(const bf16x8*)(p + 16);
  return cat8(lo, hi);
}

static __device__ inline f32x8 wmma_bf16(bf16x16 a, bf16x16 b, f32x8 c) {
  return __builtin_amdgcn_wmma_f32_16x16x32_bf16(false, a, false, b,
                                                 (short)0, c, false, false);
}

// Branch-free GEMM chunk: NT N-tiles, K-outer, A reused across tiles.
//   LDSA=false: A from global fp32 at aG (+ks*32), converted to bf16.
//   LDSA=true : A from LDS bf16 at aL (+ks*32).
//   bBase points at tile 0 of the chunk: B + (ntile0*16 + col)*ldB + half*8.
template<int NT, bool LDSA, int KSTEPS>
static __device__ inline void gemm_chunk(const float* aG, const __bf16* aL,
                                         const __bf16* bBase, int ldB,
                                         f32x8 (&acc)[NT]) {
#pragma unroll 2
  for (int ks = 0; ks < KSTEPS; ++ks) {
    bf16x16 a = LDSA ? load_a_lds(aL + ks * 32) : load_a_f32(aG + ks * 32);
#pragma unroll
    for (int t = 0; t < NT; ++t) {
      acc[t] = wmma_bf16(a, load_b(bBase + (size_t)t * 16 * ldB + ks * 32), acc[t]);
    }
  }
}

template<int NT>
static __device__ inline void zero_acc(f32x8 (&acc)[NT]) {
#pragma unroll
  for (int t = 0; t < NT; ++t)
    acc[t] = f32x8{0.f, 0.f, 0.f, 0.f, 0.f, 0.f, 0.f, 0.f};
}

// ----------------------------- packing kernels -----------------------------

__global__ void pack_vaug(const float* __restrict__ V, const float* __restrict__ gw,
                          __bf16* __restrict__ out) {
  int idx = blockIdx.x * 256 + threadIdx.x;
  if (idx >= Lc * NAUG * Dd) return;
  int i   = idx / (NAUG * Dd);
  int rem = idx % (NAUG * Dd);
  int n   = rem / Dd;
  int k   = rem % Dd;
  float v = 0.f;
  if (n < NKR) {
    int e = n >> 6, r = n & 63;
    v = V[(((size_t)i * Ec + e) * Dd + k) * Rr + r];
  } else if (n < NKR + Ec) {
    v = gw[(size_t)(n - NKR) * Dd + k];
  }
  out[idx] = (__bf16)v;
}

__global__ void pack_cblk(const float* __restrict__ C, __bf16* __restrict__ out) {
  int idx = blockIdx.x * 256 + threadIdx.x;
  if (idx >= Lc * NKR * NKR) return;
  int i   = idx / (NKR * NKR);
  int rem = idx % (NKR * NKR);
  int n   = rem / NKR;            // output rank index (r of expert e)
  int k   = rem % NKR;            // input rank index (s of expert e')
  int e   = n >> 6;
  float v = (e == (k >> 6))
          ? C[(((size_t)i * Ec + e) * Rr + (n & 63)) * Rr + (k & 63)]
          : 0.f;
  out[idx] = (__bf16)v;
}

__global__ void pack_u(const float* __restrict__ U, __bf16* __restrict__ out) {
  int idx = blockIdx.x * 256 + threadIdx.x;
  if (idx >= Lc * Dd * NKR) return;
  int i   = idx / (Dd * NKR);
  int rem = idx % (Dd * NKR);
  int n   = rem / NKR;            // output feature d
  int j   = rem % NKR;            // e*R + r
  int e   = j >> 6;
  out[idx] = (__bf16)U[(((size_t)i * Ec + e) * Dd + n) * Rr + (j & 63)];
}

// ------------------------------ main kernel --------------------------------

__global__ __launch_bounds__(WAVES * 32)
void crossnet_fused(const float* xin, const float* bias,
                    const __bf16* __restrict__ Vaug,
                    const __bf16* __restrict__ Cblk,
                    const __bf16* __restrict__ Upk,
                    float* xout) {
  __shared__ __bf16 vbuf[WAVES][TM][SV];
  __shared__ __bf16 wgbuf[WAVES][TM][SV];
  __shared__ float  gates[WAVES][TM][Ec];

  const int wave    = threadIdx.x >> 5;
  const int lane    = threadIdx.x & 31;
  const int col     = lane & 15;          // C/D column within tile; A/B row/col
  const int half    = lane >> 4;          // K-half selector
  const int rowbase = (blockIdx.x * WAVES + wave) * TM;

  for (int layer = 0; layer < Lc; ++layer) {
    const float*  xsrc = (layer == 0) ? xin : xout;   // same-wave rows only
    const __bf16* Vl = Vaug + (size_t)layer * NAUG * Dd;
    const __bf16* Cl = Cblk + (size_t)layer * NKR * NKR;
    const __bf16* Ul = Upk  + (size_t)layer * Dd * NKR;
    const float*  bl = bias + (size_t)layer * Dd;

    const float*  aRow = xsrc + (size_t)(rowbase + col) * Dd + half * 8;

    // ---- stage 1: [16 x 1024] @ [1024 x 272]; tiles 0..7, then 8..16 ----
    {
      f32x8 acc[8];
      zero_acc(acc);
      gemm_chunk<8, false, Dd / 32>(aRow, nullptr,
                                    Vl + (size_t)col * Dd + half * 8, Dd, acc);
#pragma unroll
      for (int t = 0; t < 8; ++t)
#pragma unroll
        for (int j = 0; j < 8; ++j)
          vbuf[wave][j + 8 * half][t * 16 + col] = (__bf16)tanhf(acc[t][j]);
    }
    {
      f32x8 acc[9];
      zero_acc(acc);
      gemm_chunk<9, false, Dd / 32>(aRow, nullptr,
                                    Vl + (size_t)(8 * 16 + col) * Dd + half * 8,
                                    Dd, acc);
#pragma unroll
      for (int t = 0; t < 8; ++t)
#pragma unroll
        for (int j = 0; j < 8; ++j)
          vbuf[wave][j + 8 * half][(8 + t) * 16 + col] = (__bf16)tanhf(acc[t][j]);
      if (col < Ec) {                        // tile 16: gating logits (fp32)
#pragma unroll
        for (int j = 0; j < 8; ++j)
          gates[wave][j + 8 * half][col] = acc[8][j];
      }
    }

    // ---- softmax over the 4 experts (lanes 0..15 each own one row) ----
    if (lane < TM) {
      float l0 = gates[wave][lane][0], l1 = gates[wave][lane][1];
      float l2 = gates[wave][lane][2], l3 = gates[wave][lane][3];
      float mx = fmaxf(fmaxf(l0, l1), fmaxf(l2, l3));
      float e0 = __expf(l0 - mx), e1 = __expf(l1 - mx);
      float e2 = __expf(l2 - mx), e3 = __expf(l3 - mx);
      float inv = 1.f / (e0 + e1 + e2 + e3);
      gates[wave][lane][0] = e0 * inv;
      gates[wave][lane][1] = e1 * inv;
      gates[wave][lane][2] = e2 * inv;
      gates[wave][lane][3] = e3 * inv;
    }

    // ---- stage 2: [16 x 256] @ [256 x 256] block-diag C (16 tiles) ----
#pragma unroll 1
    for (int c0 = 0; c0 < 16; c0 += 8) {
      f32x8 acc[8];
      zero_acc(acc);
      gemm_chunk<8, true, NKR / 32>(nullptr, &vbuf[wave][col][half * 8],
                                    Cl + (size_t)(c0 * 16 + col) * NKR + half * 8,
                                    NKR, acc);
#pragma unroll
      for (int t = 0; t < 8; ++t) {
        const int nt = c0 + t;
        const int e  = nt >> 2;              // expert id for this 16-col tile
#pragma unroll
        for (int j = 0; j < 8; ++j) {
          int m = j + 8 * half;
          float g = gates[wave][m][e];       // fold gate into w
          wgbuf[wave][m][nt * 16 + col] = (__bf16)(tanhf(acc[t][j]) * g);
        }
      }
    }

    // ---- stage 3: [16 x 256] @ [256 x 1024] + residual epilogue ----
#pragma unroll 1
    for (int c0 = 0; c0 < 64; c0 += 8) {
      f32x8 acc[8];
      zero_acc(acc);
      gemm_chunk<8, true, NKR / 32>(nullptr, &wgbuf[wave][col][half * 8],
                                    Ul + (size_t)(c0 * 16 + col) * NKR + half * 8,
                                    NKR, acc);
#pragma unroll
      for (int t = 0; t < 8; ++t) {
        const int d    = (c0 + t) * 16 + col;
        const float bv = bl[d];
#pragma unroll
        for (int j = 0; j < 8; ++j) {
          int gr = rowbase + j + 8 * half;
          float xlv = xsrc[(size_t)gr * Dd + d];
          float x0v = xin[(size_t)gr * Dd + d];
          xout[(size_t)gr * Dd + d] = xlv + x0v * (bv + acc[t][j]);
        }
      }
    }
    // next layer reads xout rows this wave just wrote: same-wave store->load
    // to identical addresses is kept in order (ISA 7.3); LDS deps via DScnt.
  }
}

// ------------------------------ launcher -----------------------------------

extern "C" void kernel_launch(void* const* d_in, const int* in_sizes, int n_in,
                              void* d_out, int out_size, void* d_ws, size_t ws_size,
                              hipStream_t stream) {
  const float* xin  = (const float*)d_in[0];
  const float* U    = (const float*)d_in[1];
  const float* V    = (const float*)d_in[2];
  const float* C    = (const float*)d_in[3];
  const float* gw   = (const float*)d_in[4];
  const float* bias = (const float*)d_in[5];
  float* xout = (float*)d_out;

  __bf16* Vaug = (__bf16*)d_ws;                       // 3*272*1024 elems
  __bf16* Cblk = Vaug + (size_t)Lc * NAUG * Dd;       // 3*256*256
  __bf16* Upk  = Cblk + (size_t)Lc * NKR * NKR;       // 3*1024*256

  {
    int n = Lc * NAUG * Dd;
    pack_vaug<<<(n + 255) / 256, 256, 0, stream>>>(V, gw, Vaug);
  }
  {
    int n = Lc * NKR * NKR;
    pack_cblk<<<(n + 255) / 256, 256, 0, stream>>>(C, Cblk);
  }
  {
    int n = Lc * Dd * NKR;
    pack_u<<<(n + 255) / 256, 256, 0, stream>>>(U, Upk);
  }

  const int rows_per_block = WAVES * TM;              // 128
  crossnet_fused<<<Bb / rows_per_block, WAVES * 32, 0, stream>>>(
      xin, bias, Vaug, Cblk, Upk, xout);
}
